// LatencyModel_60559038874108
// MI455X (gfx1250) — matrix operand order repaired and checked by reading, not compiled, not run on hardware
//
#include <hip/hip_runtime.h>
#include <hip/hip_bf16.h>
#include <stdint.h>

// ---------------------------------------------------------------------------
// Types (plain ext-vectors so they can live in unions)
// ---------------------------------------------------------------------------
typedef __attribute__((ext_vector_type(16))) __bf16   v16bf;
typedef __attribute__((ext_vector_type(8)))  float    v8f;
typedef __attribute__((ext_vector_type(4)))  uint32_t u32x4;
typedef __attribute__((ext_vector_type(4)))  float    f32x4;

#define NTOT 229376   // total nodes across the 4 node types
#define HD   128      // hidden width

__device__ __forceinline__ unsigned short f2bf(float f) {
  unsigned int x = __float_as_uint(f);
  x += 0x7FFFu + ((x >> 16) & 1u);          // round-to-nearest-even
  return (unsigned short)(x >> 16);
}

// ---------------------------------------------------------------------------
// f32 -> bf16, 8 elements / thread (n8 = n/8)
// ---------------------------------------------------------------------------
__global__ __launch_bounds__(256)
void k_f32_to_bf16(const float* __restrict__ s, unsigned short* __restrict__ d, int n8) {
  int i = blockIdx.x * 256 + threadIdx.x;
  if (i >= n8) return;
  const f32x4* sp = (const f32x4*)s + (size_t)i * 2;
  f32x4 a = sp[0], b = sp[1];
  u32x4 o;
  o[0] = (uint32_t)f2bf(a[0]) | ((uint32_t)f2bf(a[1]) << 16);
  o[1] = (uint32_t)f2bf(a[2]) | ((uint32_t)f2bf(a[3]) << 16);
  o[2] = (uint32_t)f2bf(b[0]) | ((uint32_t)f2bf(b[1]) << 16);
  o[3] = (uint32_t)f2bf(b[2]) | ((uint32_t)f2bf(b[3]) << 16);
  ((u32x4*)d)[i] = o;
}

// ---------------------------------------------------------------------------
// W_comb = bf16(Wlin + Wo_0 + Wo_1 + Wo_2)   (null ptrs skipped)
// ---------------------------------------------------------------------------
__global__ __launch_bounds__(256)
void k_combine_w(const float* __restrict__ w0, const float* w1, const float* w2,
                 const float* w3, unsigned short* __restrict__ out, int n) {
  int i = blockIdx.x * 256 + threadIdx.x;
  if (i >= n) return;
  float v = w0[i];
  if (w1) v += w1[i];
  if (w2) v += w2[i];
  if (w3) v += w3[i];
  out[i] = f2bf(v);
}

__global__ __launch_bounds__(128)
void k_combine_b(const float* __restrict__ b0, const float* b1, const float* b2,
                 const float* b3, float* __restrict__ out, int n) {
  int i = blockIdx.x * 128 + threadIdx.x;
  if (i >= n) return;
  float v = b0[i];
  if (b1) v += b1[i];
  if (b2) v += b2[i];
  if (b3) v += b3[i];
  out[i] = v;
}

// ---------------------------------------------------------------------------
// C[M x 128] = A[M x K](bf16) @ W[K x 128](bf16) (+ bias)   K in {64,128}
// Block: 256 thr = 8 waves; each wave owns TWO 16-row strips (32 rows) so
// every B fragment loaded from LDS feeds two v_wmma_f32_16x16x32_bf16 ops
// (halves DS traffic per matrix op).  Block tile: 256 rows x 128 cols.
// W staged transposed in LDS -> B fragments are contiguous ds_load_b128's.
// M must be a multiple of 256 (true for all call sites).
// ---------------------------------------------------------------------------
__global__ __launch_bounds__(256)
void k_gemm_bf16(const unsigned short* __restrict__ A,
                 const unsigned short* __restrict__ W,
                 const float* __restrict__ bias,
                 float* __restrict__ C, int M, int K) {
  __shared__ unsigned short Wt[128 * 128];      // Wt[c*K + k], <= 32 KB
  int tid = threadIdx.x;
  for (int i = tid; i < K * 128; i += 256) {    // cooperative transpose stage
    int k = i >> 7, c = i & 127;
    Wt[c * K + k] = W[i];
  }
  __syncthreads();

  int wave  = tid >> 5;
  int lane  = tid & 31;
  int laneM = lane & 15;
  bool hi   = lane >= 16;
  int m0    = blockIdx.x * 256 + wave * 32;     // strip 0: m0, strip 1: m0+16
  const unsigned short* arow0 = A + (size_t)(m0 + laneM) * K;
  const unsigned short* arow1 = A + (size_t)(m0 + 16 + laneM) * K;

  v8f acc[2][8];
#pragma unroll
  for (int s = 0; s < 2; ++s)
#pragma unroll
    for (int t = 0; t < 8; ++t) acc[s][t] = (v8f){0.f,0.f,0.f,0.f,0.f,0.f,0.f,0.f};

  for (int k0 = 0; k0 < K; k0 += 32) {
    // A fragments: lanes 0-15 carry K {k0..k0+7, k0+16..k0+23}; lanes 16-31 the rest
    union { v16bf v; u32x4 q[2]; } af0, af1;
    {
      const unsigned short* ap0 = arow0 + k0 + (hi ? 8 : 0);
      const unsigned short* ap1 = arow1 + k0 + (hi ? 8 : 0);
      af0.q[0] = *(const u32x4*)ap0;
      af0.q[1] = *(const u32x4*)(ap0 + 16);
      af1.q[0] = *(const u32x4*)ap1;
      af1.q[1] = *(const u32x4*)(ap1 + 16);
    }
#pragma unroll
    for (int t = 0; t < 8; ++t) {
      // B fragment: column c = t*16+laneM; lanes 0-15 -> K {k0..k0+7, k0+16..k0+23}; lanes 16-31 the rest
      union { v16bf v; u32x4 q[2]; } bf;
      const unsigned short* bp = Wt + (size_t)(t * 16 + laneM) * K + k0 + (hi ? 8 : 0);
      bf.q[0] = *(const u32x4*)bp;
      bf.q[1] = *(const u32x4*)(bp + 16);
      acc[0][t] = __builtin_amdgcn_wmma_f32_16x16x32_bf16(
          false, af0.v, false, bf.v, (short)0, acc[0][t], false, false);
      acc[1][t] = __builtin_amdgcn_wmma_f32_16x16x32_bf16(
          false, af1.v, false, bf.v, (short)0, acc[1][t], false, false);
    }
  }

  float bv[8];
#pragma unroll
  for (int t = 0; t < 8; ++t) bv[t] = bias ? bias[t * 16 + laneM] : 0.f;

#pragma unroll
  for (int s = 0; s < 2; ++s) {
    int mBase = m0 + s * 16 + (hi ? 8 : 0);
#pragma unroll
    for (int r = 0; r < 8; ++r) {
      int m = mBase + r;
      if (m < M) {
        float* cp = C + (size_t)m * HD + laneM;
#pragma unroll
        for (int t = 0; t < 8; ++t) cp[t * 16] = acc[s][t][r] + bv[t];
      }
    }
  }
}

// ---------------------------------------------------------------------------
// Edge scatter: O[dst] += Y[src]   (32 threads/edge, float4 + 4 atomics)
// ---------------------------------------------------------------------------
__global__ __launch_bounds__(256)
void k_scatter(const int* __restrict__ src, const int* __restrict__ dst,
               const float* __restrict__ Y, float* __restrict__ O, int E) {
  int gid = blockIdx.x * 256 + threadIdx.x;
  int e   = gid >> 5;
  int f   = (gid & 31) * 4;
  if (e >= E) return;
  int s = src[e], d = dst[e];
  f32x4 v = *(const f32x4*)(Y + (size_t)s * HD + f);
  float* o = O + (size_t)d * HD + f;
  atomicAdd(o + 0, v[0]);
  atomicAdd(o + 1, v[1]);
  atomicAdd(o + 2, v[2]);
  atomicAdd(o + 3, v[3]);
}

// ---------------------------------------------------------------------------
// In-place ReLU, 4 floats / thread
// ---------------------------------------------------------------------------
__global__ __launch_bounds__(256)
void k_relu(float* __restrict__ x, int n4) {
  int i = blockIdx.x * 256 + threadIdx.x;
  if (i >= n4) return;
  f32x4 v = ((f32x4*)x)[i];
#pragma unroll
  for (int k = 0; k < 4; ++k) v[k] = v[k] > 0.f ? v[k] : 0.f;
  ((f32x4*)x)[i] = v;
}

// ---------------------------------------------------------------------------
// Heads: z[n] = h[n] . W + b  packed into seq[batch, posOff + within]
// ---------------------------------------------------------------------------
__global__ __launch_bounds__(256)
void k_head(const float* __restrict__ h, const float* __restrict__ W,
            const float* __restrict__ bias, float* __restrict__ seq,
            int N, int rpb, int posOff) {
  __shared__ float ws[HD];
  if (threadIdx.x < HD) ws[threadIdx.x] = W[threadIdx.x];
  __syncthreads();
  int n = blockIdx.x * 256 + threadIdx.x;
  if (n >= N) return;
  const float* row = h + (size_t)n * HD;
  float acc = 0.f;
#pragma unroll
  for (int i = 0; i < HD; i += 4) {
    f32x4 v = *(const f32x4*)(row + i);
    acc += v[0] * ws[i] + v[1] * ws[i + 1] + v[2] * ws[i + 2] + v[3] * ws[i + 3];
  }
  acc += bias[0];
  int bb = n / rpb, w = n % rpb;
  seq[(size_t)bb * 896 + posOff + w] = acc;
}

// ---------------------------------------------------------------------------
// LSTM: one workgroup per batch element, thread j owns hidden unit j.
// Whh is repacked in LDS as float4 per (m, j): the 4 gate columns of unit j,
// so the 896-step serial loop does one ds_load_b128 per m instead of 4xb32.
// Ping-pong h buffer -> one barrier per step.
// ---------------------------------------------------------------------------
__global__ __launch_bounds__(64)
void k_lstm(const float* __restrict__ seq, const float* __restrict__ Wih,
            const float* __restrict__ Whh, const float* __restrict__ bih,
            const float* __restrict__ bhh, const float* __restrict__ Wout,
            const float* __restrict__ bout, float* __restrict__ out) {
  __shared__ f32x4 Ws4[64 * 64];      // Ws4[m*64 + j] = {Wi, Wf, Wg, Wo} of unit j
  __shared__ float hbuf[2][64];
  __shared__ float red[64];
  int j = threadIdx.x, bb = blockIdx.x;
  for (int m = 0; m < 64; ++m) {
    f32x4 w;
#pragma unroll
    for (int k = 0; k < 4; ++k) w[k] = Whh[m * 256 + k * 64 + j];
    Ws4[m * 64 + j] = w;
  }
  float wi[4], bs[4];
#pragma unroll
  for (int k = 0; k < 4; ++k) {
    wi[k] = Wih[k * 64 + j];
    bs[k] = bih[k * 64 + j] + bhh[k * 64 + j];
  }
  hbuf[0][j] = 0.f;
  float c = 0.f;
  __syncthreads();
  int p = 0;
  const float* sq = seq + (size_t)bb * 896;
  for (int t = 0; t < 896; ++t) {
    float xt = sq[t];
    float g[4];
#pragma unroll
    for (int k = 0; k < 4; ++k) g[k] = fmaf(xt, wi[k], bs[k]);
    for (int m = 0; m < 64; ++m) {
      float hm = hbuf[p][m];
      f32x4 w = Ws4[m * 64 + j];
#pragma unroll
      for (int k = 0; k < 4; ++k) g[k] = fmaf(hm, w[k], g[k]);
    }
    float ig = 1.f / (1.f + __expf(-g[0]));
    float fg = 1.f / (1.f + __expf(-g[1]));
    float gg = tanhf(g[2]);
    float og = 1.f / (1.f + __expf(-g[3]));
    c = fg * c + ig * gg;
    hbuf[p ^ 1][j] = og * tanhf(c);
    __syncthreads();
    p ^= 1;
  }
  red[j] = hbuf[p][j] * Wout[j];
  __syncthreads();
  if (j == 0) {
    float s = 0.f;
    for (int m = 0; m < 64; ++m) s += red[m];
    out[bb] = s + bout[0];
  }
}

// ---------------------------------------------------------------------------
// Host orchestration
// ---------------------------------------------------------------------------
extern "C" void kernel_launch(void* const* d_in, const int* in_sizes, int n_in,
                              void* d_out, int out_size, void* d_ws, size_t ws_size,
                              hipStream_t stream) {
  (void)in_sizes; (void)n_in; (void)out_size; (void)ws_size;

  // node types: 0=start_task 1=task 2=link 3=end_task
  static const int ntOff[4]   = {0, 16384, 81920, 212992};
  static const int ntCnt[4]   = {16384, 65536, 131072, 16384};
  static const int ntAlpha[4] = {2, 3, 1, 0};      // -> sorted-key position
  static const int posOff[4]  = {0, 64, 320, 832}; // seq pack offsets
  // edges in ETS (data) order
  static const int eSrc[7]   = {0, 1, 1, 2, 2, 2, 3};
  static const int eDst[7]   = {1, 0, 2, 1, 2, 3, 2};
  static const int eCnt[7]   = {65536, 65536, 524288, 524288, 1048576, 65536, 65536};
  static const int eAlpha[7] = {4, 6, 5, 3, 2, 1, 0}; // -> sorted-key position
  static const int inE[4][3] = {{1,-1,-1}, {0,3,-1}, {2,4,6}, {5,-1,-1}};

  // params pytree flattened with sorted dict keys:
  // conv(3x7x[Wo,Wr,br])=63, head(4x[W,b])=8, lin(3x4x[W,b])=24,
  // lstm[Whh,Wih,bhh,bih]=4, out[W,b]=2
  const int PB = 15;
  auto convWo = [&](int l, int e) { return (const float*)d_in[PB + l * 21 + eAlpha[e] * 3 + 0]; };
  auto convWr = [&](int l, int e) { return (const float*)d_in[PB + l * 21 + eAlpha[e] * 3 + 1]; };
  auto convBr = [&](int l, int e) { return (const float*)d_in[PB + l * 21 + eAlpha[e] * 3 + 2]; };
  auto headW  = [&](int nt)       { return (const float*)d_in[PB + 63 + ntAlpha[nt] * 2 + 0]; };
  auto headB  = [&](int nt)       { return (const float*)d_in[PB + 63 + ntAlpha[nt] * 2 + 1]; };
  auto linW   = [&](int l, int nt){ return (const float*)d_in[PB + 71 + l * 8 + ntAlpha[nt] * 2 + 0]; };
  auto linB   = [&](int l, int nt){ return (const float*)d_in[PB + 71 + l * 8 + ntAlpha[nt] * 2 + 1]; };
  const float* Whh  = (const float*)d_in[PB + 95];
  const float* Wih  = (const float*)d_in[PB + 96];
  const float* bhh  = (const float*)d_in[PB + 97];
  const float* bih  = (const float*)d_in[PB + 98];
  const float* Wout = (const float*)d_in[PB + 99];
  const float* bout = (const float*)d_in[PB + 100];

  // workspace carve-up (~362 MB total)
  char* wp = (char*)d_ws;
  auto carve = [&](size_t bytes) { void* p = wp; wp += (bytes + 255) & ~(size_t)255; return p; };
  float*          hA    = (float*)carve(sizeof(float) * (size_t)NTOT * HD);
  float*          hB    = (float*)carve(sizeof(float) * (size_t)NTOT * HD);
  unsigned short* hb    = (unsigned short*)carve(2ull * NTOT * HD);
  float*          y     = (float*)carve(sizeof(float) * 131072ull * HD);
  unsigned short* wcomb = (unsigned short*)carve(2ull * 4 * 128 * HD);
  float*          bcomb = (float*)carve(sizeof(float) * 4 * HD);
  unsigned short* wrb   = (unsigned short*)carve(2ull * 7 * 128 * HD);
  float*          seq   = (float*)carve(sizeof(float) * 256 * 896);

  for (int l = 0; l < 3; ++l) {
    const int fi = (l == 0) ? 64 : 128;
    const float* cur = (l == 1) ? hB : hA;       // layer 1 reads hB, layer 2 reads hA
    float* outO      = (l == 1) ? hA : hB;       // l0->hB, l1->hA, l2->hB

    // a) current activations -> bf16
    if (l == 0) {
      for (int nt = 0; nt < 4; ++nt) {
        int n8 = ntCnt[nt] * 64 / 8;
        k_f32_to_bf16<<<n8 / 256, 256, 0, stream>>>(
            (const float*)d_in[nt], hb + (size_t)ntOff[nt] * 64, n8);
      }
    } else {
      int n8 = NTOT * HD / 8;
      k_f32_to_bf16<<<n8 / 256, 256, 0, stream>>>(cur, hb, n8);
    }

    // b) fold lin + root weights:  W_comb = W_lin + sum(Wo),  b_comb = b + sum(br)
    for (int nt = 0; nt < 4; ++nt) {
      const float *w1 = nullptr, *w2 = nullptr, *w3 = nullptr;
      const float *b1 = nullptr, *b2 = nullptr, *b3 = nullptr;
      int c = 0;
      for (int k = 0; k < 3; ++k) {
        int e = inE[nt][k];
        if (e < 0) break;
        if      (c == 0) { w1 = convWo(l, e); b1 = convBr(l, e); }
        else if (c == 1) { w2 = convWo(l, e); b2 = convBr(l, e); }
        else             { w3 = convWo(l, e); b3 = convBr(l, e); }
        ++c;
      }
      int n = fi * HD;
      k_combine_w<<<(n + 255) / 256, 256, 0, stream>>>(
          linW(l, nt), w1, w2, w3, wcomb + (size_t)nt * fi * HD, n);
      k_combine_b<<<1, 128, 0, stream>>>(linB(l, nt), b1, b2, b3, bcomb + nt * HD, HD);
    }
    for (int e = 0; e < 7; ++e) {
      int n8 = fi * HD / 8;
      k_f32_to_bf16<<<(n8 + 255) / 256, 256, 0, stream>>>(
          convWr(l, e), wrb + (size_t)e * fi * HD, n8);
    }

    // c) self GEMMs:  o = h @ W_comb + b_comb  (writes the whole o buffer)
    for (int nt = 0; nt < 4; ++nt) {
      k_gemm_bf16<<<ntCnt[nt] / 256, 256, 0, stream>>>(
          hb + (size_t)ntOff[nt] * fi, wcomb + (size_t)nt * fi * HD,
          bcomb + nt * HD, outO + (size_t)ntOff[nt] * HD, ntCnt[nt], fi);
    }

    // d) per edge type: y = h_src @ Wr  then  o[dst] += y[src]
    for (int e = 0; e < 7; ++e) {
      int sN = ntCnt[eSrc[e]];
      k_gemm_bf16<<<sN / 256, 256, 0, stream>>>(
          hb + (size_t)ntOff[eSrc[e]] * fi, wrb + (size_t)e * fi * HD,
          nullptr, y, sN, fi);
      const int* ei = (const int*)d_in[4 + e];
      k_scatter<<<eCnt[e] / 8, 256, 0, stream>>>(
          ei, ei + eCnt[e], y, outO + (size_t)ntOff[eDst[e]] * HD, eCnt[e]);
    }

    // e) ReLU after layers 0 and 1
    if (l < 2) {
      int n4 = NTOT * HD / 4;
      k_relu<<<n4 / 256, 256, 0, stream>>>(outO, n4);
    }
  }

  // heads -> packed sequence  (final h is in hB)
  for (int nt = 0; nt < 4; ++nt) {
    k_head<<<ntCnt[nt] / 256, 256, 0, stream>>>(
        hB + (size_t)ntOff[nt] * HD, headW(nt), headB(nt), seq,
        ntCnt[nt], ntCnt[nt] / 256, posOff[nt]);
  }

  // LSTM + output projection
  k_lstm<<<256, 64, 0, stream>>>(seq, Wih, Whh, bih, bhh, Wout, bout, (float*)d_out);
}
